// SolvGNNCat_36189394437141
// MI455X (gfx1250) — compile-verified, matrix operand order checked by reading, not run on hardware
//
#include <hip/hip_runtime.h>
#include <math.h>

// ---------------- problem constants ----------------
constexpr int kB   = 4096;
constexpr int kNPG = 20;
constexpr int kN   = kB * kNPG;    // 81920
constexpr int kE   = kN * 2;       // 163840
constexpr int kIN  = 74;
constexpr int kH   = 128;
constexpr int kEH  = 32;
constexpr int k2B  = 2 * kB;       // 8192
constexpr int kE4  = 4 * kB;       // 16384
constexpr int kKNN = kEH * kH + kH;  // 4224
constexpr int kLDnf  = 132;        // padded node_feat ld (even, > 131)
constexpr int kLDxg  = 260;        // padded xgf ld (even, > 257)

typedef __attribute__((ext_vector_type(16))) __bf16 v16bf;
typedef __attribute__((ext_vector_type(8)))  __bf16 v8bf;
typedef __attribute__((ext_vector_type(8)))  float  v8f;

// ---------------- fragment loaders ----------------
// A fragment (16x32 bf16) from fp32 row-major, fully in-bounds fast path.
// Per ISA: lane 0-15 row=lane K={0..7,16..23}; lane 16-31 row=lane-16 K={8..15,24..31}
__device__ __forceinline__ v16bf a_vec(const float* __restrict__ p) {
  v16bf f;
#pragma unroll
  for (int jj = 0; jj < 4; ++jj) {
    float2 q0 = *reinterpret_cast<const float2*>(p + 2 * jj);
    float2 q1 = *reinterpret_cast<const float2*>(p + 16 + 2 * jj);
    f[2 * jj]         = (__bf16)q0.x;  f[2 * jj + 1]     = (__bf16)q0.y;
    f[8 + 2 * jj]     = (__bf16)q1.x;  f[8 + 2 * jj + 1] = (__bf16)q1.y;
  }
  return f;
}
// guarded scalar A fragment for the K tail
__device__ __forceinline__ v16bf a_tail(const float* __restrict__ A, int lda,
                                        int m0, int k0, int K, int lane) {
  const int r  = lane & 15;
  const int kb = (lane >> 4) << 3;
  const float* row = A + (size_t)(m0 + r) * lda;
  v16bf f;
#pragma unroll
  for (int j = 0; j < 8; ++j) {
    int k = k0 + kb + ((j & 4) << 2) + ((j & 3) << 1);
    f[2 * j]     = (k     < K) ? (__bf16)row[k]     : (__bf16)0.0f;
    f[2 * j + 1] = (k + 1 < K) ? (__bf16)row[k + 1] : (__bf16)0.0f;
  }
  return f;
}
// B fragment (32x16 bf16) from packed bf16 [N][kpad]: two b128 loads
__device__ __forceinline__ v16bf b_vec(const __bf16* __restrict__ p) {
  v8bf lo = *reinterpret_cast<const v8bf*>(p);        // K  .. K+7
  v8bf hi = *reinterpret_cast<const v8bf*>(p + 16);   // K+16..K+23
  return __builtin_shufflevector(lo, hi, 0, 1, 2, 3, 4, 5, 6, 7,
                                 8, 9, 10, 11, 12, 13, 14, 15);
}

// ---------------- generic bf16 WMMA GEMM: 16x64 strip per wave ----------------
template <bool RELU>
__global__ void gemm_wmma(const float* __restrict__ A, int lda,
                          const __bf16* __restrict__ Bp, int kpad,
                          const float* __restrict__ bias,
                          const float* __restrict__ add,
                          float* __restrict__ C, int ldc,
                          int M, int Nn, int K) {
  const int lane   = threadIdx.x & 31;
  const int wave   = blockIdx.x * (blockDim.x >> 5) + (threadIdx.x >> 5);
  const int tilesN = Nn >> 6;
  const int tiles  = (M >> 4) * tilesN;
  if (wave >= tiles) return;                    // wave-uniform guard
  const int m0 = (wave / tilesN) << 4;
  const int n0 = (wave % tilesN) << 6;
  const int kb = (lane >> 4) << 3;
  const float* rowA = A + (size_t)(m0 + (lane & 15)) * lda;
  const __bf16* rb0 = Bp + (size_t)(n0 +  0 + (lane & 15)) * kpad;
  const __bf16* rb1 = Bp + (size_t)(n0 + 16 + (lane & 15)) * kpad;
  const __bf16* rb2 = Bp + (size_t)(n0 + 32 + (lane & 15)) * kpad;
  const __bf16* rb3 = Bp + (size_t)(n0 + 48 + (lane & 15)) * kpad;
  v8f acc0 = (v8f){0,0,0,0,0,0,0,0}, acc1 = acc0, acc2 = acc0, acc3 = acc0;
  const int Kmain = K & ~31;
  for (int k0 = 0; k0 < Kmain; k0 += 32) {
    if (k0 + 32 < Kmain)                        // -> global_prefetch_b8
      __builtin_prefetch(rowA + k0 + 32, 0, 0);
    v16bf a = a_vec(rowA + k0 + kb);
    v16bf b0 = b_vec(rb0 + k0 + kb);
    v16bf b1 = b_vec(rb1 + k0 + kb);
    v16bf b2 = b_vec(rb2 + k0 + kb);
    v16bf b3 = b_vec(rb3 + k0 + kb);
    acc0 = __builtin_amdgcn_wmma_f32_16x16x32_bf16(false, a, false, b0, (short)0, acc0, false, false);
    acc1 = __builtin_amdgcn_wmma_f32_16x16x32_bf16(false, a, false, b1, (short)0, acc1, false, false);
    acc2 = __builtin_amdgcn_wmma_f32_16x16x32_bf16(false, a, false, b2, (short)0, acc2, false, false);
    acc3 = __builtin_amdgcn_wmma_f32_16x16x32_bf16(false, a, false, b3, (short)0, acc3, false, false);
  }
  if (Kmain < K) {                              // B is zero-padded to kpad
    v16bf a = a_tail(A, lda, m0, Kmain, K, lane);
    v16bf b0 = b_vec(rb0 + Kmain + kb);
    v16bf b1 = b_vec(rb1 + Kmain + kb);
    v16bf b2 = b_vec(rb2 + Kmain + kb);
    v16bf b3 = b_vec(rb3 + Kmain + kb);
    acc0 = __builtin_amdgcn_wmma_f32_16x16x32_bf16(false, a, false, b0, (short)0, acc0, false, false);
    acc1 = __builtin_amdgcn_wmma_f32_16x16x32_bf16(false, a, false, b1, (short)0, acc1, false, false);
    acc2 = __builtin_amdgcn_wmma_f32_16x16x32_bf16(false, a, false, b2, (short)0, acc2, false, false);
    acc3 = __builtin_amdgcn_wmma_f32_16x16x32_bf16(false, a, false, b3, (short)0, acc3, false, false);
  }
  const int n  = lane & 15;
  const int mb = (lane >> 4) << 3;
  v8f accs[4] = {acc0, acc1, acc2, acc3};
#pragma unroll
  for (int t = 0; t < 4; ++t) {
    const int nc = n0 + 16 * t + n;
    const float bv = bias ? bias[nc] : 0.0f;
#pragma unroll
    for (int r = 0; r < 8; ++r) {
      const int m = m0 + mb + r;
      float v = accs[t][r] + bv;
      if (add) v += add[(size_t)m * ldc + nc];
      if (RELU) v = v > 0.0f ? v : 0.0f;
      C[(size_t)m * ldc + nc] = v;
    }
  }
}

// ---- fused NNConv GEMM: m[e,:] = kron(g_e, h0_src) @ [We2;be2mat]  (K=4224)
__global__ void nnconv_gemm(const float* __restrict__ g,    // [16384, 32]
                            const float* __restrict__ h0,   // [8192, 128]
                            const __bf16* __restrict__ Bp,  // packed [128][4224]
                            float* __restrict__ mo) {       // [16384, 128]
  const int lane = threadIdx.x & 31;
  const int wave = blockIdx.x * (blockDim.x >> 5) + (threadIdx.x >> 5);
  const int tiles = (kE4 >> 4) * 2;             // 16x64 strips, Nn=128
  if (wave >= tiles) return;
  const int e0 = (wave >> 1) << 4;
  const int n0 = (wave & 1) << 6;
  const int er = e0 + (lane & 15);
  const int s  = (er < k2B) ? er : er - k2B;
  const float* h0row = h0 + (size_t)s * kH;
  const float* grow  = g + (size_t)er * kEH;
  const int kb = (lane >> 4) << 3;
  // preload this lane's h0 slices for the 4 phases (fragment order)
  float hv[4][16];
#pragma unroll
  for (int p = 0; p < 4; ++p) {
    const float* hp = h0row + p * 32 + kb;
#pragma unroll
    for (int jj = 0; jj < 4; ++jj) {
      float2 q0 = *reinterpret_cast<const float2*>(hp + 2 * jj);
      float2 q1 = *reinterpret_cast<const float2*>(hp + 16 + 2 * jj);
      hv[p][2 * jj] = q0.x;      hv[p][2 * jj + 1] = q0.y;
      hv[p][8 + 2 * jj] = q1.x;  hv[p][8 + 2 * jj + 1] = q1.y;
    }
  }
  const __bf16* rb0 = Bp + (size_t)(n0 +  0 + (lane & 15)) * kKNN;
  const __bf16* rb1 = Bp + (size_t)(n0 + 16 + (lane & 15)) * kKNN;
  const __bf16* rb2 = Bp + (size_t)(n0 + 32 + (lane & 15)) * kKNN;
  const __bf16* rb3 = Bp + (size_t)(n0 + 48 + (lane & 15)) * kKNN;
  v8f acc0 = (v8f){0,0,0,0,0,0,0,0}, acc1 = acc0, acc2 = acc0, acc3 = acc0;
  for (int kk = 0; kk < kEH; ++kk) {            // main: P-block, K=4096
    const float gk = grow[kk];
#pragma unroll
    for (int p = 0; p < 4; ++p) {
      const int k0 = kk * 128 + p * 32;
      v16bf a;
#pragma unroll
      for (int i = 0; i < 16; ++i) a[i] = (__bf16)(gk * hv[p][i]);
      v16bf b0 = b_vec(rb0 + k0 + kb);
      v16bf b1 = b_vec(rb1 + k0 + kb);
      v16bf b2 = b_vec(rb2 + k0 + kb);
      v16bf b3 = b_vec(rb3 + k0 + kb);
      acc0 = __builtin_amdgcn_wmma_f32_16x16x32_bf16(false, a, false, b0, (short)0, acc0, false, false);
      acc1 = __builtin_amdgcn_wmma_f32_16x16x32_bf16(false, a, false, b1, (short)0, acc1, false, false);
      acc2 = __builtin_amdgcn_wmma_f32_16x16x32_bf16(false, a, false, b2, (short)0, acc2, false, false);
      acc3 = __builtin_amdgcn_wmma_f32_16x16x32_bf16(false, a, false, b3, (short)0, acc3, false, false);
    }
  }
#pragma unroll
  for (int p = 0; p < 4; ++p) {                 // tail: be2-block, K=128
    const int k0 = kEH * kH + p * 32;
    v16bf a;
#pragma unroll
    for (int i = 0; i < 16; ++i) a[i] = (__bf16)hv[p][i];
    v16bf b0 = b_vec(rb0 + k0 + kb);
    v16bf b1 = b_vec(rb1 + k0 + kb);
    v16bf b2 = b_vec(rb2 + k0 + kb);
    v16bf b3 = b_vec(rb3 + k0 + kb);
    acc0 = __builtin_amdgcn_wmma_f32_16x16x32_bf16(false, a, false, b0, (short)0, acc0, false, false);
    acc1 = __builtin_amdgcn_wmma_f32_16x16x32_bf16(false, a, false, b1, (short)0, acc1, false, false);
    acc2 = __builtin_amdgcn_wmma_f32_16x16x32_bf16(false, a, false, b2, (short)0, acc2, false, false);
    acc3 = __builtin_amdgcn_wmma_f32_16x16x32_bf16(false, a, false, b3, (short)0, acc3, false, false);
  }
  const int n  = lane & 15;
  const int mb = (lane >> 4) << 3;
  v8f accs[4] = {acc0, acc1, acc2, acc3};
#pragma unroll
  for (int t = 0; t < 4; ++t)
#pragma unroll
    for (int r = 0; r < 8; ++r)
      mo[(size_t)(e0 + mb + r) * kH + n0 + 16 * t + n] = accs[t][r];
}

// ---------------- weight packing: fp32 -> bf16 [N][kpad] ----------------
__global__ void k_pack_b(const float* __restrict__ src, __bf16* __restrict__ dst,
                         int K, int Nn, int kpad) {       // src [K x N] row-major
  int t = blockIdx.x * blockDim.x + threadIdx.x;
  if (t >= Nn * kpad) return;
  int n = t / kpad, k = t % kpad;
  dst[t] = (k < K) ? (__bf16)src[(size_t)k * Nn + n] : (__bf16)0.0f;
}
__global__ void k_pack_bt(const float* __restrict__ src, __bf16* __restrict__ dst,
                          int K, int Nn, int kpad) {      // src [N x K] row-major
  int t = blockIdx.x * blockDim.x + threadIdx.x;
  if (t >= Nn * kpad) return;
  int n = t / kpad, k = t % kpad;
  dst[t] = (k < K) ? (__bf16)src[(size_t)n * K + k] : (__bf16)0.0f;
}
__global__ void k_pack_we2(const float* __restrict__ We2, const float* __restrict__ be2,
                           __bf16* __restrict__ dst) {    // -> [128][4224]
  int t = blockIdx.x * blockDim.x + threadIdx.x;
  if (t >= kH * kKNN) return;
  int n = t / kKNN, k = t % kKNN;
  float v = (k < kEH * kH) ? We2[(size_t)k * kH + n] : be2[(size_t)(k - kEH * kH) * kH + n];
  dst[t] = (__bf16)v;
}

// ---------------- GCN support kernels ----------------
__global__ void k_fill1(float* __restrict__ p, int n) {
  int i = blockIdx.x * blockDim.x + threadIdx.x;
  if (i < n) p[i] = 1.0f;
}
__global__ void k_deg_edges(const int* __restrict__ ei, float* __restrict__ deg) {
  int e = blockIdx.x * blockDim.x + threadIdx.x;
  if (e < kE) atomicAdd(&deg[ei[kE + e]], 1.0f);
}
__global__ void k_dinv(float* __restrict__ deg, int n) {
  int i = blockIdx.x * blockDim.x + threadIdx.x;
  if (i < n) deg[i] = rsqrtf(deg[i]);
}
__global__ void k_agg_init(const float* __restrict__ h, const float* __restrict__ dinv,
                           float* __restrict__ agg) {
  size_t t = (size_t)blockIdx.x * blockDim.x + threadIdx.x;
  if (t >= (size_t)kN * kH) return;
  float d = dinv[(int)(t >> 7)];
  agg[t] = d * d * h[t];                       // self-loop term
}
__global__ void k_agg_edges(const int* __restrict__ ei, const float* __restrict__ h,
                            const float* __restrict__ dinv, float* __restrict__ agg) {
  int t = blockIdx.x * blockDim.x + threadIdx.x;   // kE * 32 threads, 4 feats each
  if (t >= kE * (kH / 4)) return;
  int e = t >> 5, f4 = (t & 31) << 2;
  int s = ei[e], d = ei[kE + e];
  float w = dinv[s] * dinv[d];
  float4 hv = *reinterpret_cast<const float4*>(h + (size_t)s * kH + f4);
  float* ap = agg + (size_t)d * kH + f4;
  atomicAdd(ap + 0, hv.x * w);
  atomicAdd(ap + 1, hv.y * w);
  atomicAdd(ap + 2, hv.z * w);
  atomicAdd(ap + 3, hv.w * w);
}
__global__ void k_bias_relu(float* __restrict__ x, const float* __restrict__ b) {
  size_t t = (size_t)blockIdx.x * blockDim.x + threadIdx.x;
  if (t >= (size_t)kN * kH) return;
  float v = x[t] + b[t & 127];
  x[t] = v > 0.0f ? v : 0.0f;
}
__global__ void k_pool(const float* __restrict__ x, float* __restrict__ xg) {
  int t = blockIdx.x * blockDim.x + threadIdx.x;
  if (t >= kB * kH) return;
  int b = t >> 7, f = t & 127;
  float s = 0.0f;
#pragma unroll
  for (int p = 0; p < kNPG; ++p) s += x[(size_t)(b * kNPG + p) * kH + f];
  xg[t] = s * (1.0f / (float)kNPG);
}

// ---------------- MPNN support kernels ----------------
__global__ void k_node_feat(const float* __restrict__ xg1, const float* __restrict__ xg2,
                            const float* __restrict__ ap, const float* __restrict__ bp,
                            const float* __restrict__ tps1, const float* __restrict__ tps2,
                            float* __restrict__ nf) {      // [8192][kLDnf]
  int t = blockIdx.x * blockDim.x + threadIdx.x;
  if (t >= k2B * (kH + 3)) return;
  int i = t / (kH + 3), c = t % (kH + 3);
  int ib = (i < kB) ? i : i - kB;
  float v;
  if (c < kH)            v = (i < kB) ? xg1[(size_t)i * kH + c] : xg2[(size_t)ib * kH + c];
  else if (c == kH)      v = ap[ib];
  else if (c == kH + 1)  v = bp[ib];
  else                   v = (i < kB) ? tps1[ib] : tps2[ib];
  nf[(size_t)i * kLDnf + c] = v;
}
__global__ void k_edge_gate(const float* __restrict__ hb1, const float* __restrict__ hb2,
                            const float* __restrict__ We1, const float* __restrict__ be1,
                            float* __restrict__ g) {
  int t = blockIdx.x * blockDim.x + threadIdx.x;
  if (t >= kE4 * kEH) return;
  int e = t >> 5, k = t & 31;
  float ef = (e < 3 * kB) ? hb1[e % kB] : hb2[e - 3 * kB];
  float v = ef * We1[k] + be1[k];
  g[t] = v > 0.0f ? v : 0.0f;
}
__global__ void k_aggr(const float* __restrict__ m, float* __restrict__ aggr) {
  int t = blockIdx.x * blockDim.x + threadIdx.x;
  if (t >= k2B * kH) return;
  int i = t >> 7, f = t & 127;
  float v;
  if (i < kB) v = m[(size_t)(kB + i) * kH + f] + m[(size_t)(2 * kB + i) * kH + f];
  else { int ii = i - kB; v = m[(size_t)ii * kH + f] + m[(size_t)(3 * kB + ii) * kH + f]; }
  aggr[t] = v;
}
__device__ __forceinline__ float sigmoidf_(float x) { return 1.0f / (1.0f + expf(-x)); }
__global__ void k_gru(const float* __restrict__ gi, const float* __restrict__ gh,
                      const float* __restrict__ h0, float* __restrict__ hout) {
  int t = blockIdx.x * blockDim.x + threadIdx.x;
  if (t >= k2B * kH) return;
  int i = t >> 7, f = t & 127;
  const float* gir = gi + (size_t)i * 3 * kH;
  const float* ghr = gh + (size_t)i * 3 * kH;
  float r = sigmoidf_(gir[f] + ghr[f]);
  float z = sigmoidf_(gir[kH + f] + ghr[kH + f]);
  float nn = tanhf(gir[2 * kH + f] + r * ghr[2 * kH + f]);
  hout[t] = (1.0f - z) * nn + z * h0[t];
}
__global__ void k_xgf(const float* __restrict__ h, const float* __restrict__ Tx,
                      float* __restrict__ xgf) {           // [4096][kLDxg]
  int t = blockIdx.x * blockDim.x + threadIdx.x;
  if (t >= kB * (2 * kH + 1)) return;
  int b = t / (2 * kH + 1), c = t % (2 * kH + 1);
  float v;
  if (c < kH)          v = h[(size_t)b * kH + c];
  else if (c < 2 * kH) v = h[(size_t)(kB + b) * kH + (c - kH)];
  else                 v = (Tx[b] + 60.0f) / 349.3f;
  xgf[(size_t)b * kLDxg + c] = v;
}
__global__ void k_final(const float* __restrict__ o2, const float* __restrict__ Wm3,
                        const float* __restrict__ bm3, float* __restrict__ out) {
  int b = blockIdx.x * blockDim.x + threadIdx.x;
  if (b >= kB) return;
  float s = bm3[0];
#pragma unroll 8
  for (int k = 0; k < kH; ++k) s += o2[(size_t)b * kH + k] * Wm3[k];
  out[b] = s;
}

// ---------------- host side ----------------
static inline int cdiv(long long a, long long b) { return (int)((a + b - 1) / b); }

extern "C" void kernel_launch(void* const* d_in, const int* in_sizes, int n_in,
                              void* d_out, int out_size, void* d_ws, size_t ws_size,
                              hipStream_t stream) {
  (void)in_sizes; (void)n_in; (void)out_size; (void)ws_size;
  const float* solvent_x = (const float*)d_in[0];
  const int*   solv_ei   = (const int*)d_in[1];
  const float* solute_x  = (const float*)d_in[3];
  const int*   solu_ei   = (const int*)d_in[4];
  const float* ap        = (const float*)d_in[6];
  const float* bp        = (const float*)d_in[7];
  const float* tps1      = (const float*)d_in[8];
  const float* tps2      = (const float*)d_in[9];
  const float* hb1       = (const float*)d_in[10];
  const float* hb2       = (const float*)d_in[11];
  const float* Tx        = (const float*)d_in[12];
  const float* W_gcn1    = (const float*)d_in[13];
  const float* b_gcn1    = (const float*)d_in[14];
  const float* W_gcn2    = (const float*)d_in[15];
  const float* b_gcn2    = (const float*)d_in[16];
  const float* Wp        = (const float*)d_in[17];
  const float* bp_proj   = (const float*)d_in[18];
  const float* We1       = (const float*)d_in[19];
  const float* be1       = (const float*)d_in[20];
  const float* We2       = (const float*)d_in[21];
  const float* be2       = (const float*)d_in[22];
  const float* W_root    = (const float*)d_in[23];
  const float* b_nn      = (const float*)d_in[24];
  const float* W_ih      = (const float*)d_in[25];
  const float* W_hh      = (const float*)d_in[26];
  const float* b_ih      = (const float*)d_in[27];
  const float* b_hh      = (const float*)d_in[28];
  const float* Wm1       = (const float*)d_in[29];
  const float* bm1       = (const float*)d_in[30];
  const float* Wm2       = (const float*)d_in[31];
  const float* bm2       = (const float*)d_in[32];
  const float* Wm3       = (const float*)d_in[33];
  const float* bm3       = (const float*)d_in[34];
  float* out = (float*)d_out;

  // ---- workspace layout ----
  float* ws   = (float*)d_ws;
  float* big0 = ws;                               // N*H
  float* big1 = ws + (size_t)kN * kH;             // N*H
  float* dinv = ws + 2ull * kN * kH;              // N
  float* xg1  = dinv + kN;                        // B*H
  float* xg2  = xg1 + (size_t)kB * kH;            // B*H
  // packed bf16 weights after the fp32 region
  __bf16* pk = (__bf16*)(xg2 + (size_t)kB * kH);
  __bf16* pWg1 = pk;  pk += 128 * 96;             // W_gcn1  K=74  ->96
  __bf16* pWg2 = pk;  pk += 128 * 128;            // W_gcn2
  __bf16* pWpj = pk;  pk += 128 * 160;            // Wp      K=131->160
  __bf16* pWrt = pk;  pk += 128 * 128;            // W_root
  __bf16* pWih = pk;  pk += 384 * 128;            // W_ih (already N-major)
  __bf16* pWhh = pk;  pk += 384 * 128;            // W_hh
  __bf16* pWm1 = pk;  pk += 256 * 288;            // Wm1     K=257->288
  __bf16* pWm2 = pk;  pk += 128 * 256;            // Wm2
  __bf16* pWee = pk;  pk += 128 * kKNN;           // [We2; be2mat]
  // big0 repartition for MPNN stage
  float* m_buf = big0;                                        // 16384*128
  float* gi    = m_buf + (size_t)kE4 * kH;                    // 8192*384
  float* gh    = gi + (size_t)k2B * 3 * kH;                   // 8192*384
  float* aggr  = gh + (size_t)k2B * 3 * kH;                   // 8192*128
  // big1 repartition
  float* nfeat = big1;                                        // 8192*kLDnf
  float* h0    = nfeat + (size_t)k2B * kLDnf;                 // 8192*128
  float* gbuf  = h0 + (size_t)k2B * kH;                       // 16384*32
  float* nf    = gbuf + (size_t)kE4 * kEH;                    // 8192*128
  float* hfin  = nf + (size_t)k2B * kH;                       // 8192*128
  float* xgf   = hfin + (size_t)k2B * kH;                     // 4096*kLDxg
  float* o1    = xgf + (size_t)kB * kLDxg;                    // 4096*256
  float* o2    = o1 + (size_t)kB * 2 * kH;                    // 4096*128

  const int T = 256;
  const size_t NH = (size_t)kN * kH;
  auto gb = [](int M, int Nn) { return cdiv((long long)(M >> 4) * (Nn >> 6), 8); };

  // ---- pack weights (bf16, N-major, K padded to 32) ----
  k_pack_b <<<cdiv(128 * 96, T), T, 0, stream>>>(W_gcn1, pWg1, kIN, kH, 96);
  k_pack_b <<<cdiv(128 * 128, T), T, 0, stream>>>(W_gcn2, pWg2, kH, kH, 128);
  k_pack_b <<<cdiv(128 * 160, T), T, 0, stream>>>(Wp, pWpj, kH + 3, kH, 160);
  k_pack_b <<<cdiv(128 * 128, T), T, 0, stream>>>(W_root, pWrt, kH, kH, 128);
  k_pack_bt<<<cdiv(384 * 128, T), T, 0, stream>>>(W_ih, pWih, kH, 3 * kH, 128);
  k_pack_bt<<<cdiv(384 * 128, T), T, 0, stream>>>(W_hh, pWhh, kH, 3 * kH, 128);
  k_pack_b <<<cdiv(256 * 288, T), T, 0, stream>>>(Wm1, pWm1, 2 * kH + 1, 2 * kH, 288);
  k_pack_b <<<cdiv(128 * 256, T), T, 0, stream>>>(Wm2, pWm2, 2 * kH, kH, 256);
  k_pack_we2<<<cdiv(kH * kKNN, T), T, 0, stream>>>(We2, be2, pWee);

  // ================= GCN stacks (solvent then solute, shared weights) =====
  for (int mol = 0; mol < 2; ++mol) {
    const float* x  = mol ? solute_x : solvent_x;
    const int*   ei = mol ? solu_ei : solv_ei;
    float*       xg = mol ? xg2 : xg1;
    k_fill1<<<cdiv(kN, T), T, 0, stream>>>(dinv, kN);
    k_deg_edges<<<cdiv(kE, T), T, 0, stream>>>(ei, dinv);
    k_dinv<<<cdiv(kN, T), T, 0, stream>>>(dinv, kN);
    gemm_wmma<false><<<gb(kN, kH), T, 0, stream>>>(
        x, kIN, pWg1, 96, nullptr, nullptr, big0, kH, kN, kH, kIN);
    k_agg_init<<<cdiv((long long)NH, T), T, 0, stream>>>(big0, dinv, big1);
    k_agg_edges<<<cdiv(kE * 32, T), T, 0, stream>>>(ei, big0, dinv, big1);
    k_bias_relu<<<cdiv((long long)NH, T), T, 0, stream>>>(big1, b_gcn1);
    gemm_wmma<false><<<gb(kN, kH), T, 0, stream>>>(
        big1, kH, pWg2, 128, nullptr, nullptr, big0, kH, kN, kH, kH);
    k_agg_init<<<cdiv((long long)NH, T), T, 0, stream>>>(big0, dinv, big1);
    k_agg_edges<<<cdiv(kE * 32, T), T, 0, stream>>>(ei, big0, dinv, big1);
    k_bias_relu<<<cdiv((long long)NH, T), T, 0, stream>>>(big1, b_gcn2);
    k_pool<<<cdiv(kB * kH, T), T, 0, stream>>>(big1, xg);
  }

  // ================= MPNN stage ==========================================
  k_node_feat<<<cdiv(k2B * (kH + 3), T), T, 0, stream>>>(
      xg1, xg2, ap, bp, tps1, tps2, nfeat);
  gemm_wmma<true><<<gb(k2B, kH), T, 0, stream>>>(
      nfeat, kLDnf, pWpj, 160, bp_proj, nullptr, h0, kH, k2B, kH, kH + 3);
  k_edge_gate<<<cdiv(kE4 * kEH, T), T, 0, stream>>>(hb1, hb2, We1, be1, gbuf);
  nnconv_gemm<<<cdiv((long long)(kE4 >> 4) * 2, 8), T, 0, stream>>>(
      gbuf, h0, pWee, m_buf);
  k_aggr<<<cdiv(k2B * kH, T), T, 0, stream>>>(m_buf, aggr);
  gemm_wmma<true><<<gb(k2B, kH), T, 0, stream>>>(
      h0, kH, pWrt, 128, b_nn, aggr, nf, kH, k2B, kH, kH);
  gemm_wmma<false><<<gb(k2B, 3 * kH), T, 0, stream>>>(
      nf, kH, pWih, 128, b_ih, nullptr, gi, 3 * kH, k2B, 3 * kH, kH);
  gemm_wmma<false><<<gb(k2B, 3 * kH), T, 0, stream>>>(
      h0, kH, pWhh, 128, b_hh, nullptr, gh, 3 * kH, k2B, 3 * kH, kH);
  k_gru<<<cdiv(k2B * kH, T), T, 0, stream>>>(gi, gh, h0, hfin);
  k_xgf<<<cdiv(kB * (2 * kH + 1), T), T, 0, stream>>>(hfin, Tx, xgf);
  gemm_wmma<true><<<gb(kB, 2 * kH), T, 0, stream>>>(
      xgf, kLDxg, pWm1, 288, bm1, nullptr, o1, 2 * kH, kB, 2 * kH, 2 * kH + 1);
  gemm_wmma<true><<<gb(kB, kH), T, 0, stream>>>(
      o1, 2 * kH, pWm2, 256, bm2, nullptr, o2, kH, kB, kH, 2 * kH);
  k_final<<<cdiv(kB, T), T, 0, stream>>>(o2, Wm3, bm3, out);
}
